// InfiniAttention_4277787427206
// MI455X (gfx1250) — compile-verified, matrix operand order checked by reading, not run on hardware
//
#include <hip/hip_runtime.h>

typedef __attribute__((ext_vector_type(16))) _Float16 v16h;
typedef __attribute__((ext_vector_type(8)))  _Float16 v8h;
typedef __attribute__((ext_vector_type(8)))  float    v8f;
typedef __attribute__((ext_vector_type(4)))  int      v4i;

#define EMBED 768
#define NHEAD 12
#define HDIM  64
#define MSIZE 64
#define BATCH 8
#define SEQ   1024
#define MROWS (BATCH*SEQ)   /* 8192 */
#define QKVN  (3*EMBED)     /* 2304 */

#if defined(__gfx1250__) && __has_builtin(__builtin_amdgcn_global_load_async_to_lds_b128)
#define HAVE_ASYNC_LDS 1
#else
#define HAVE_ASYNC_LDS 0
#endif

// ---------------------------------------------------------------------------
// CDNA5 async global->LDS copy (16B per lane), tracked by ASYNCcnt.
// Builtin signature: (v4i addrspace(1)*, v4i addrspace(3)*, imm offset, imm cpol)
// ---------------------------------------------------------------------------
__device__ __forceinline__ void copy16_to_lds(_Float16* lds, const _Float16* g) {
#if HAVE_ASYNC_LDS
  __builtin_amdgcn_global_load_async_to_lds_b128(
      (__attribute__((address_space(1))) v4i*)g,
      (__attribute__((address_space(3))) v4i*)lds, 0, 0);
#else
  *(uint4*)lds = *(const uint4*)g;   // sync fallback: load + ds_store
#endif
}

__device__ __forceinline__ void async_copies_wait() {
#if HAVE_ASYNC_LDS
#if __has_builtin(__builtin_amdgcn_s_wait_asynccnt)
  __builtin_amdgcn_s_wait_asynccnt(0);
#else
  asm volatile("s_wait_asynccnt 0" ::: "memory");
#endif
#endif
}

// ---------------------------------------------------------------------------
// WMMA helpers (CDNA5 16x16x32 f16 -> f32)
// ---------------------------------------------------------------------------
__device__ __forceinline__ v8f wmma16(v16h a, v16h b, v8f c) {
  return __builtin_amdgcn_wmma_f32_16x16x32_f16(false, a, false, b, (short)0, c,
                                                false, false);
}

__device__ __forceinline__ v8f vzero() {
  v8f z;
#pragma unroll
  for (int i = 0; i < 8; ++i) z[i] = 0.f;
  return z;
}

// 16x32 (MxK) A-style fragment (also B^T given N-major storage).
// ISA 7.12.2: halves 0-7 <-> K kc+half*8..+7, halves 8-15 <-> +16. Row = lane%16.
__device__ __forceinline__ v16h load_frag(const _Float16* base, int ld, int row,
                                          int kc, int half) {
  const _Float16* p = base + (size_t)row * ld + kc + half * 8;
  v8h lo = *(const v8h*)p;
  v8h hi = *(const v8h*)(p + 16);
  return __builtin_shufflevector(lo, hi, 0, 1, 2, 3, 4, 5, 6, 7, 8, 9, 10, 11,
                                 12, 13, 14, 15);
}

// Cross-lane reductions over the 16-lane half-group holding one C-row.
__device__ __forceinline__ float red_max16(float v) {
  v = fmaxf(v, __shfl_xor(v, 1, 32));
  v = fmaxf(v, __shfl_xor(v, 2, 32));
  v = fmaxf(v, __shfl_xor(v, 4, 32));
  v = fmaxf(v, __shfl_xor(v, 8, 32));
  return v;
}
__device__ __forceinline__ float red_sum16(float v) {
  v += __shfl_xor(v, 1, 32);
  v += __shfl_xor(v, 2, 32);
  v += __shfl_xor(v, 4, 32);
  v += __shfl_xor(v, 8, 32);
  return v;
}

// ---------------------------------------------------------------------------
// Precision / layout conversion kernels
// ---------------------------------------------------------------------------
__global__ void cvt_f16(const float* __restrict__ in, _Float16* __restrict__ out,
                        int n) {
  int i = blockIdx.x * blockDim.x + threadIdx.x;
  if (i < n) out[i] = (_Float16)in[i];
}

// in: (K x N) row-major fp32  ->  out: (N x K) row-major f16  (B^T layout)
__global__ void transpose_cvt(const float* __restrict__ in,
                              _Float16* __restrict__ out, int K, int N) {
  int i = blockIdx.x * blockDim.x + threadIdx.x;
  if (i >= K * N) return;
  int n = i / K, k = i % K;
  out[i] = (_Float16)in[(size_t)k * N + n];
}

// memory(64,768) @ w_mem(768,768) + b_mem in exact fp32; emit per-head f16
// layouts: memk (H,M,D) for Q@K^T and memvT (H,D,M) for P@V.
__global__ void mem_project(const float* __restrict__ memory,
                            const float* __restrict__ w_mem,
                            const float* __restrict__ b_mem,
                            _Float16* __restrict__ memk,
                            _Float16* __restrict__ memvT) {
  int i = blockIdx.x * blockDim.x + threadIdx.x;
  if (i >= MSIZE * EMBED) return;
  int m = i / EMBED, c = i % EMBED;
  const float* xr = memory + (size_t)m * EMBED;
  float acc = b_mem[c];
  for (int k = 0; k < EMBED; ++k) acc += xr[k] * w_mem[(size_t)k * EMBED + c];
  int h = c >> 6, d = c & 63;
  _Float16 hv = (_Float16)acc;
  memk[((size_t)h * MSIZE + m) * HDIM + d] = hv;
  memvT[((size_t)h * HDIM + d) * MSIZE + m] = hv;
}

// ---------------------------------------------------------------------------
// QKV projection GEMM: xh(8192x768) @ wqkvT^T + b_qkv, scatter to Q/K/V^T.
// One wave = 16(M) x 64(N) tile; block = 4 waves (64 M-rows) sharing the
// B^T tile, which is double-buffered in LDS via async global->LDS DMA.
// ---------------------------------------------------------------------------
__global__ void qkv_gemm(const _Float16* __restrict__ A,
                         const _Float16* __restrict__ Bt,
                         const float* __restrict__ bias,
                         _Float16* __restrict__ qh, _Float16* __restrict__ kh,
                         _Float16* __restrict__ vTh) {
  __shared__ __align__(16) _Float16 Bbuf[2][64 * 32];
  const int tid = threadIdx.x;
  const int wave = tid >> 5, lane = tid & 31;
  const int l16 = lane & 15, half = lane >> 4;
  const int m0 = blockIdx.y * 64 + wave * 16;
  const int n0 = blockIdx.x * 64;

  // B-tile staging: 64x32 f16 = 256 16B-chunks; 128 threads x 2 chunks.
  const int c0 = tid, c1 = tid + 128;
  const int r0 = c0 >> 2, o0 = (c0 & 3) * 8;
  const int r1 = c1 >> 2, o1 = (c1 & 3) * 8;

  auto stage_b = [&](int kc, int buf) {
    copy16_to_lds(&Bbuf[buf][r0 * 32 + o0],
                  Bt + (size_t)(n0 + r0) * EMBED + kc + o0);
    copy16_to_lds(&Bbuf[buf][r1 * 32 + o1],
                  Bt + (size_t)(n0 + r1) * EMBED + kc + o1);
  };

  v8f acc[4];
#pragma unroll
  for (int j = 0; j < 4; ++j) acc[j] = vzero();

  stage_b(0, 0);
  int buf = 0;
  for (int kc = 0; kc < EMBED; kc += 32) {
    async_copies_wait();
    __syncthreads();
    if (kc + 32 < EMBED) stage_b(kc + 32, buf ^ 1);
    __builtin_prefetch(A + (size_t)(m0 + l16) * EMBED + kc + 64, 0, 3);
    v16h a = load_frag(A, EMBED, m0 + l16, kc, half);
#pragma unroll
    for (int j = 0; j < 4; ++j) {
      v16h bf = load_frag(&Bbuf[buf][0], 32, j * 16 + l16, 0, half);
      acc[j] = wmma16(a, bf, acc[j]);
    }
    buf ^= 1;
  }

#pragma unroll
  for (int j = 0; j < 4; ++j) {
    const int n = n0 + j * 16 + l16;
    const float bv = bias[n];
    const int which = n / EMBED;       // 0=q 1=k 2=v (uniform per 16-col tile)
    const int c = n % EMBED;
    const int h = c >> 6, d = c & 63;
#pragma unroll
    for (int r = 0; r < 8; ++r) {
      const int m = m0 + half * 8 + r;
      const int b = m >> 10, t = m & (SEQ - 1);
      const _Float16 hv = (_Float16)(acc[j][r] + bv);
      const size_t bh = (size_t)b * NHEAD + h;
      if (which == 0)      qh[(bh * SEQ + t) * HDIM + d] = hv;
      else if (which == 1) kh[(bh * SEQ + t) * HDIM + d] = hv;
      else                 vTh[(bh * HDIM + d) * SEQ + t] = hv;
    }
  }
}

// ---------------------------------------------------------------------------
// Output projection GEMM: attnh(8192x768) @ woutT^T + b_out -> fp32 out.
// Same LDS-staged structure as qkv_gemm.
// ---------------------------------------------------------------------------
__global__ void out_gemm(const _Float16* __restrict__ A,
                         const _Float16* __restrict__ Bt,
                         const float* __restrict__ bias,
                         float* __restrict__ out) {
  __shared__ __align__(16) _Float16 Bbuf[2][64 * 32];
  const int tid = threadIdx.x;
  const int wave = tid >> 5, lane = tid & 31;
  const int l16 = lane & 15, half = lane >> 4;
  const int m0 = blockIdx.y * 64 + wave * 16;
  const int n0 = blockIdx.x * 64;

  const int c0 = tid, c1 = tid + 128;
  const int r0 = c0 >> 2, o0 = (c0 & 3) * 8;
  const int r1 = c1 >> 2, o1 = (c1 & 3) * 8;

  auto stage_b = [&](int kc, int buf) {
    copy16_to_lds(&Bbuf[buf][r0 * 32 + o0],
                  Bt + (size_t)(n0 + r0) * EMBED + kc + o0);
    copy16_to_lds(&Bbuf[buf][r1 * 32 + o1],
                  Bt + (size_t)(n0 + r1) * EMBED + kc + o1);
  };

  v8f acc[4];
#pragma unroll
  for (int j = 0; j < 4; ++j) acc[j] = vzero();

  stage_b(0, 0);
  int buf = 0;
  for (int kc = 0; kc < EMBED; kc += 32) {
    async_copies_wait();
    __syncthreads();
    if (kc + 32 < EMBED) stage_b(kc + 32, buf ^ 1);
    __builtin_prefetch(A + (size_t)(m0 + l16) * EMBED + kc + 64, 0, 3);
    v16h a = load_frag(A, EMBED, m0 + l16, kc, half);
#pragma unroll
    for (int j = 0; j < 4; ++j) {
      v16h bf = load_frag(&Bbuf[buf][0], 32, j * 16 + l16, 0, half);
      acc[j] = wmma16(a, bf, acc[j]);
    }
    buf ^= 1;
  }
#pragma unroll
  for (int j = 0; j < 4; ++j) {
    const int n = n0 + j * 16 + l16;
    const float bv = bias[n];
#pragma unroll
    for (int r = 0; r < 8; ++r) {
      const int m = m0 + half * 8 + r;
      out[(size_t)m * EMBED + n] = acc[j][r] + bv;
    }
  }
}

// ---------------------------------------------------------------------------
// Flash-attention softmax step (shared by local + memory passes).
// ---------------------------------------------------------------------------
__device__ __forceinline__ v16h softmax_step(v8f S0, v8f S1, float scale,
                                             float (&mstat)[8], float (&lstat)[8],
                                             v8f (&O)[4], _Float16* myP,
                                             int l16, int half) {
#pragma unroll
  for (int r = 0; r < 8; ++r) {
    float s0 = S0[r] * scale, s1 = S1[r] * scale;
    float rm = red_max16(fmaxf(s0, s1));
    float mn = fmaxf(mstat[r], rm);
    float al = __expf(mstat[r] - mn);
    float p0 = __expf(s0 - mn);
    float p1 = __expf(s1 - mn);
    lstat[r] = lstat[r] * al + red_sum16(p0 + p1);
    mstat[r] = mn;
    O[0][r] *= al; O[1][r] *= al; O[2][r] *= al; O[3][r] *= al;
    myP[(half * 8 + r) * 32 + l16] = (_Float16)p0;
    myP[(half * 8 + r) * 32 + 16 + l16] = (_Float16)p1;
  }
  // Per-wave LDS transpose fence: DS ops are in-order within a wave.
  asm volatile("s_wait_dscnt 0" ::: "memory");
  const _Float16* pp = myP + l16 * 32 + half * 8;
  v8h lo = *(const v8h*)pp;
  v8h hi = *(const v8h*)(pp + 16);
  return __builtin_shufflevector(lo, hi, 0, 1, 2, 3, 4, 5, 6, 7, 8, 9, 10, 11,
                                 12, 13, 14, 15);
}

// ---------------------------------------------------------------------------
// Attention: 8 waves/block, all sharing one (b,h); one wave per 16-query tile.
// K and V^T tiles are double-buffered in LDS via async DMA and shared by all
// 8 waves. Local softmax over 1024 keys + independent softmax over 64 memory
// keys; outputs summed and stored f16 for the output projection.
// ---------------------------------------------------------------------------
__global__ void attn_kernel(const _Float16* __restrict__ qh,
                            const _Float16* __restrict__ kh,
                            const _Float16* __restrict__ vTh,
                            const _Float16* __restrict__ memk,
                            const _Float16* __restrict__ memvT,
                            _Float16* __restrict__ attnh) {
  __shared__ __align__(16) _Float16 Kbuf[2][32 * 64];  // keys x D
  __shared__ __align__(16) _Float16 Vbuf[2][64 * 32];  // D x keys
  __shared__ __align__(16) _Float16 Pt[8][16 * 32];    // per-wave P tiles
  const int tid = threadIdx.x;
  const int wave = tid >> 5, lane = tid & 31;
  const int l16 = lane & 15, half = lane >> 4;
  const int tile = blockIdx.x * 8 + wave;              // 0..6143
  const int q0 = (tile & 63) << 4;
  const int bh = tile >> 6;                            // same for whole block
  const int h = bh % NHEAD, b = bh / NHEAD;

  const _Float16* Q   = qh  + (size_t)bh * SEQ * HDIM;
  const _Float16* K   = kh  + (size_t)bh * SEQ * HDIM;
  const _Float16* VT  = vTh + (size_t)bh * HDIM * SEQ;
  const _Float16* MK  = memk  + (size_t)h * MSIZE * HDIM;
  const _Float16* MVT = memvT + (size_t)h * HDIM * MSIZE;
  _Float16* myP = &Pt[wave][0];

  // Tile staging: K tile 32x64 and V^T tile 64x32, each 256 16B-chunks over
  // 256 threads -> one chunk of each per thread.
  const int kr = tid >> 3, ko = (tid & 7) * 8;   // K: row=key, col=d
  const int vr = tid >> 2, vo = (tid & 3) * 8;   // V^T: row=d, col=key

  auto stage_kv = [&](int kb, int buf) {
    copy16_to_lds(&Kbuf[buf][kr * 64 + ko], K + (size_t)(kb + kr) * HDIM + ko);
    copy16_to_lds(&Vbuf[buf][vr * 32 + vo], VT + (size_t)vr * SEQ + kb + vo);
  };

  const v16h qf0 = load_frag(Q, HDIM, q0 + l16, 0, half);
  const v16h qf1 = load_frag(Q, HDIM, q0 + l16, 32, half);
  const float scale = 0.125f;                          // 1/sqrt(64)

  float mstat[8], lstat[8];
  v8f O[4];
#pragma unroll
  for (int r = 0; r < 8; ++r) { mstat[r] = -__builtin_inff(); lstat[r] = 0.f; }
#pragma unroll
  for (int j = 0; j < 4; ++j) O[j] = vzero();

  // ---- local attention over SEQ keys, 32 per iteration, double-buffered ----
  stage_kv(0, 0);
  int buf = 0;
  for (int kb = 0; kb < SEQ; kb += 32) {
    async_copies_wait();
    __syncthreads();
    if (kb + 32 < SEQ) stage_kv(kb + 32, buf ^ 1);
    const _Float16* Kb = &Kbuf[buf][0];
    v8f S0 = vzero(), S1 = vzero();
    S0 = wmma16(qf0, load_frag(Kb, 64, l16, 0, half), S0);
    S0 = wmma16(qf1, load_frag(Kb, 64, l16, 32, half), S0);
    S1 = wmma16(qf0, load_frag(Kb, 64, 16 + l16, 0, half), S1);
    S1 = wmma16(qf1, load_frag(Kb, 64, 16 + l16, 32, half), S1);
    v16h pf = softmax_step(S0, S1, scale, mstat, lstat, O, myP, l16, half);
    const _Float16* Vb = &Vbuf[buf][0];
#pragma unroll
    for (int j = 0; j < 4; ++j)
      O[j] = wmma16(pf, load_frag(Vb, 32, j * 16 + l16, 0, half), O[j]);
    buf ^= 1;
  }
  v8f Ofin[4];
#pragma unroll
  for (int j = 0; j < 4; ++j) {
#pragma unroll
    for (int r = 0; r < 8; ++r) Ofin[j][r] = O[j][r] / lstat[r];
  }

  // ---- memory attention over MSIZE keys (independent softmax, tiny) ----
#pragma unroll
  for (int r = 0; r < 8; ++r) { mstat[r] = -__builtin_inff(); lstat[r] = 0.f; }
#pragma unroll
  for (int j = 0; j < 4; ++j) O[j] = vzero();
  for (int kb = 0; kb < MSIZE; kb += 32) {
    v8f S0 = vzero(), S1 = vzero();
    S0 = wmma16(qf0, load_frag(MK, HDIM, kb + l16, 0, half), S0);
    S0 = wmma16(qf1, load_frag(MK, HDIM, kb + l16, 32, half), S0);
    S1 = wmma16(qf0, load_frag(MK, HDIM, kb + 16 + l16, 0, half), S1);
    S1 = wmma16(qf1, load_frag(MK, HDIM, kb + 16 + l16, 32, half), S1);
    v16h pf = softmax_step(S0, S1, scale, mstat, lstat, O, myP, l16, half);
#pragma unroll
    for (int j = 0; j < 4; ++j)
      O[j] = wmma16(pf, load_frag(MVT, MSIZE, j * 16 + l16, kb, half), O[j]);
  }
#pragma unroll
  for (int j = 0; j < 4; ++j) {
#pragma unroll
    for (int r = 0; r < 8; ++r) Ofin[j][r] += O[j][r] / lstat[r];
  }

  // ---- store as f16 in (B*T, EMBED) for the output projection ----
#pragma unroll
  for (int j = 0; j < 4; ++j) {
#pragma unroll
    for (int r = 0; r < 8; ++r) {
      const int row = b * SEQ + q0 + half * 8 + r;
      const int col = h * HDIM + j * 16 + l16;
      attnh[(size_t)row * EMBED + col] = (_Float16)Ofin[j][r];
    }
  }
}

// ---------------------------------------------------------------------------
// Host launcher
// ---------------------------------------------------------------------------
extern "C" void kernel_launch(void* const* d_in, const int* in_sizes, int n_in,
                              void* d_out, int out_size, void* d_ws,
                              size_t ws_size, hipStream_t stream) {
  const float* x      = (const float*)d_in[0];
  const float* w_qkv  = (const float*)d_in[1];
  const float* b_qkv  = (const float*)d_in[2];
  const float* w_out  = (const float*)d_in[3];
  const float* b_out  = (const float*)d_in[4];
  const float* w_mem  = (const float*)d_in[5];
  const float* b_mem  = (const float*)d_in[6];
  const float* memory = (const float*)d_in[7];
  float* out = (float*)d_out;

  char* p = (char*)d_ws;
  auto take = [&](size_t bytes) {
    char* r = p;
    p += (bytes + 255) & ~(size_t)255;
    return r;
  };
  _Float16* xh    = (_Float16*)take((size_t)MROWS * EMBED * 2);
  _Float16* wqkvT = (_Float16*)take((size_t)QKVN * EMBED * 2);
  _Float16* woutT = (_Float16*)take((size_t)EMBED * EMBED * 2);
  _Float16* qh    = (_Float16*)take((size_t)BATCH * NHEAD * SEQ * HDIM * 2);
  _Float16* kh    = (_Float16*)take((size_t)BATCH * NHEAD * SEQ * HDIM * 2);
  _Float16* vTh   = (_Float16*)take((size_t)BATCH * NHEAD * HDIM * SEQ * 2);
  _Float16* memk  = (_Float16*)take((size_t)NHEAD * MSIZE * HDIM * 2);
  _Float16* memvT = (_Float16*)take((size_t)NHEAD * HDIM * MSIZE * 2);
  _Float16* attnh = (_Float16*)take((size_t)MROWS * EMBED * 2);

  const int nX = MROWS * EMBED;
  cvt_f16<<<(nX + 255) / 256, 256, 0, stream>>>(x, xh, nX);
  transpose_cvt<<<(EMBED * QKVN + 255) / 256, 256, 0, stream>>>(w_qkv, wqkvT,
                                                                EMBED, QKVN);
  transpose_cvt<<<(EMBED * EMBED + 255) / 256, 256, 0, stream>>>(w_out, woutT,
                                                                 EMBED, EMBED);
  mem_project<<<(MSIZE * EMBED + 255) / 256, 256, 0, stream>>>(memory, w_mem,
                                                               b_mem, memk,
                                                               memvT);
  qkv_gemm<<<dim3(QKVN / 64, MROWS / 64), 128, 0, stream>>>(xh, wqkvT, b_qkv,
                                                            qh, kh, vTh);
  attn_kernel<<<(BATCH * NHEAD * (SEQ / 16)) / 8, 256, 0, stream>>>(
      qh, kh, vTh, memk, memvT, attnh);
  out_gemm<<<dim3(EMBED / 64, MROWS / 64), 128, 0, stream>>>(attnh, woutT,
                                                             b_out, out);
}
